// GravitationalAttentionHead_32993938768206
// MI455X (gfx1250) — matrix operand order repaired and checked by reading, not compiled, not run on hardware
//
#include <hip/hip_runtime.h>

#define B_ 4
#define S_ 4096
#define D_ 64
#define J_ 4          // j-range split: 4x more waves, partials are additive (fixed shift)
#define EVT 1e-6f
#define MAXF 50.0f
#define CURV 0.15f

typedef __attribute__((ext_vector_type(16))) _Float16 v16h;
typedef __attribute__((ext_vector_type(8)))  float    v8f;

// A-fragment K map (16-bit A 16x32): halfs 0-7 -> K 0-7(+8*lhalf), halfs 8-15 -> K 16-23(+8*lhalf)
__device__ __forceinline__ int kmap(int h, int lhalf) {
    return (h & 7) + ((h >> 3) << 4) + (lhalf << 3);
}

// ---------------------------------------------------------------------------
// Kernel 1: masses[b,s] = softplus(x[b,s,:] . mass_w) -> ws table + output[1].
// ---------------------------------------------------------------------------
__global__ void k_masses(const float* __restrict__ x,
                         const float* __restrict__ mw,
                         float* __restrict__ mG,
                         float* __restrict__ out_m) {
    int idx = blockIdx.x * blockDim.x + threadIdx.x;   // b*S + s
    const float* xr = x + (size_t)idx * D_;
    float z = 0.f;
#pragma unroll
    for (int d = 0; d < D_; ++d) z = fmaf(xr[d], mw[d], z);
    float m = (z > 20.f) ? z : log1pf(__expf(z));      // softplus
    mG[idx]   = m;
    out_m[idx] = m;
}

// ---------------------------------------------------------------------------
// Kernel 2: pre-swizzle x (f32) into f16 WMMA B-fragment layout.
// xB flat index = ((b*128 + jt)*4 + dt)*32 + lane, 16 halfs each:
//   half h = x[b][jt*32 + kmap(h, lane>>4)][dt*16 + (lane&15)]
// ---------------------------------------------------------------------------
__global__ void k_pack(const float* __restrict__ x, v16h* __restrict__ xB) {
    int t = blockIdx.x * blockDim.x + threadIdx.x;     // 65536 total
    int lane = t & 31;
    int dt   = (t >> 5) & 3;
    int jt   = (t >> 7) & 127;
    int b    = t >> 14;
    int jbase = jt * 32;
    int dcol  = dt * 16 + (lane & 15);
    v16h v;
#pragma unroll
    for (int h = 0; h < 16; ++h)
        v[h] = (_Float16)x[((size_t)b * S_ + jbase + kmap(h, lane >> 4)) * D_ + dcol];
    xB[t] = v;
}

// ---------------------------------------------------------------------------
// Kernel 3: batch-invariant geometry table, computed ONCE (not per batch):
//   R[i,j] = rcp(max(d2*(1+CURV*cos(sqrt(d2+EVT))), EVT))
// stored pre-swizzled in A-fragment order: Rsw[((itile*128+jt)*32+lane)*16+h],
// i = itile*16 + (lane&15), j = jt*32 + kmap(h, lane>>4).  64 MB -> lives in L2.
// ---------------------------------------------------------------------------
__global__ void k_geo(const float* __restrict__ pos, float* __restrict__ Rsw) {
    int t = blockIdx.x * blockDim.x + threadIdx.x;     // 256*128*32 = 1,048,576
    int lane  = t & 31;
    int jt    = (t >> 5) & 127;
    int itile = t >> 12;
    int i = itile * 16 + (lane & 15);
    int lhalf = lane >> 4;
    const float px = pos[i * 3 + 0];
    const float py = pos[i * 3 + 1];
    const float pz = pos[i * 3 + 2];
    float r[16];
#pragma unroll
    for (int h = 0; h < 16; ++h) {
        int j = jt * 32 + kmap(h, lhalf);
        float dx = pos[j * 3 + 0] - px;
        float dy = pos[j * 3 + 1] - py;
        float dz = pos[j * 3 + 2] - pz;
        float d2 = fmaf(dx, dx, fmaf(dy, dy, dz * dz));
        float dn = __builtin_sqrtf(d2 + EVT);
        float d2m = d2 * fmaf(CURV, __cosf(dn), 1.0f);
        d2m = fmaxf(d2m, EVT);
        r[h] = __builtin_amdgcn_rcpf(d2m);
    }
    float4* o = (float4*)(Rsw + (size_t)t * 16);
#pragma unroll
    for (int q = 0; q < 4; ++q)
        o[q] = make_float4(r[4 * q], r[4 * q + 1], r[4 * q + 2], r[4 * q + 3]);
}

// ---------------------------------------------------------------------------
// Kernel 4: flash loop. Grid = B * 64 iblocks * J_ jsplits = 1024 blocks,
// 4 waves/block (16 rows each) -> 4096 waves. Fixed softmax shift (row max
// is statically MAX_FORCE via the clamped diagonal) makes j-partials additive:
// each wave writes unnormalized acc + row-sum partials (no atomics).
// ---------------------------------------------------------------------------
__global__ void __launch_bounds__(128) k_attn(const v16h*  __restrict__ xB,
                                              const float* __restrict__ Rsw,
                                              const float* __restrict__ mG,
                                              const float* __restrict__ G,
                                              float* __restrict__ pacc,
                                              float* __restrict__ pl) {
    __shared__ float ms[S_];                           // 16 KB of 320 KB WGP LDS
    const int iblk  = blockIdx.x & 63;
    const int js    = (blockIdx.x >> 6) & (J_ - 1);
    const int batch = blockIdx.x >> 8;
    const int tid   = threadIdx.x;

    {   // stage masses for this batch (vectorized)
        float4* ms4 = (float4*)ms;
        const float4* mg4 = (const float4*)(mG + batch * S_);
        for (int k = tid; k < S_ / 4; k += 128) ms4[k] = mg4[k];
    }
    __syncthreads();

    const int wave  = tid >> 5;
    const int lane  = tid & 31;
    const int lhalf = lane >> 4;
    const int i0    = iblk * 64 + wave * 16;
    const int itile = (i0 >> 4);                       // iblk*4 + wave
    const int i     = i0 + (lane & 15);

    const float Gmi = fabsf(G[0]) * ms[i];

    v8f acc0 = {}, acc1 = {}, acc2 = {}, acc3 = {};
    float lsum = 0.f;

    const v16h*  xBb = xB + ((size_t)batch * 128 * 4 * 32);
    const int jbeg = js * (S_ / J_);
    const int jend = jbeg + (S_ / J_);

    for (int j0 = jbeg; j0 < jend; j0 += 32) {
        const int jt = j0 >> 5;
        const float4* rp = (const float4*)(Rsw + (((size_t)itile * 128 + jt) * 32 + lane) * 16);
        if (j0 + 32 < jend) {                          // CDNA5 global_prefetch_b8
            __builtin_prefetch((const void*)(rp + 128), 0, 3);
            __builtin_prefetch((const void*)(xBb + (jt + 1) * 128 + lane), 0, 3);
        }

        // ---- geometry (batch-invariant, from L2) ----
        float rr[16];
        {
            float4 q0 = rp[0], q1 = rp[1], q2 = rp[2], q3 = rp[3];
            rr[0]=q0.x; rr[1]=q0.y; rr[2]=q0.z; rr[3]=q0.w;
            rr[4]=q1.x; rr[5]=q1.y; rr[6]=q1.z; rr[7]=q1.w;
            rr[8]=q2.x; rr[9]=q2.y; rr[10]=q2.z; rr[11]=q2.w;
            rr[12]=q3.x; rr[13]=q3.y; rr[14]=q3.z; rr[15]=q3.w;
        }

        // ---- scores: p = exp(min(G*mi*mj*R, 50) - 50), A-fragment order ----
        v16h a;
        float psum = 0.f;
#pragma unroll
        for (int h = 0; h < 16; ++h) {
            const int j = j0 + kmap(h, lhalf);
            float f = Gmi * ms[j] * rr[h];
            f = fminf(f, MAXF);
            const float p = __expf(f - MAXF);
            psum += p;
            a[h] = (_Float16)p;
        }
        lsum += psum;

        // ---- V tiles: one 32B vector load each (2x global_load_b128) ----
        const v16h* bp = xBb + (size_t)jt * 128 + lane;
        const v16h b0 = bp[0];
        const v16h b1 = bp[32];
        const v16h b2 = bp[64];
        const v16h b3 = bp[96];

        acc0 = __builtin_amdgcn_wmma_f32_16x16x32_f16(false, a, false, b0, (short)0, acc0, false, false);
        acc1 = __builtin_amdgcn_wmma_f32_16x16x32_f16(false, a, false, b1, (short)0, acc1, false, false);
        acc2 = __builtin_amdgcn_wmma_f32_16x16x32_f16(false, a, false, b2, (short)0, acc2, false, false);
        acc3 = __builtin_amdgcn_wmma_f32_16x16x32_f16(false, a, false, b3, (short)0, acc3, false, false);
    }

    // Row m's probabilities live in lanes m and m+16 (A layout) -> pair reduce.
    const float lrow = lsum + __shfl_xor(lsum, 16, 32);
    const size_t part = (size_t)(js * B_ + batch) * S_;
    if (lane < 16) pl[part + i0 + lane] = lrow;

    const int nn = lane & 15;
#pragma unroll
    for (int r = 0; r < 8; ++r) {
        const int row = r + (lhalf << 3);              // C layout: VGPR r <-> row r+8*lhalf
        const size_t o = (part + i0 + row) * D_ + nn;
        pacc[o +  0] = acc0[r];
        pacc[o + 16] = acc1[r];
        pacc[o + 32] = acc2[r];
        pacc[o + 48] = acc3[r];
    }
}

// ---------------------------------------------------------------------------
// Kernel 5: combine j-partials: out = (sum_js acc) / (sum_js l)
// ---------------------------------------------------------------------------
__global__ void k_combine(const float* __restrict__ pacc,
                          const float* __restrict__ pl,
                          float* __restrict__ out) {
    int t = blockIdx.x * blockDim.x + threadIdx.x;     // B*S*D = 1,048,576
    int bi = t >> 6;                                   // b*S + i
    float l = 0.f, v = 0.f;
#pragma unroll
    for (int js = 0; js < J_; ++js) {
        l += pl[(size_t)js * B_ * S_ + bi];
        v += pacc[(size_t)js * B_ * S_ * D_ + t];
    }
    out[t] = v / l;
}

// ---------------------------------------------------------------------------
extern "C" void kernel_launch(void* const* d_in, const int* in_sizes, int n_in,
                              void* d_out, int out_size, void* d_ws, size_t ws_size,
                              hipStream_t stream) {
    const float* x   = (const float*)d_in[0];   // (4, 4096, 64)
    const float* pos = (const float*)d_in[1];   // (4096, 3)
    const float* G   = (const float*)d_in[2];   // scalar
    const float* mw  = (const float*)d_in[3];   // (1, 64)

    float* out   = (float*)d_out;                        // (4,4096,64)
    float* out_m = out + (size_t)B_ * S_ * D_;           // (4,4096)

    // workspace layout
    char* ws = (char*)d_ws;
    float* mG  = (float*)ws;                                           //  64 KB
    v16h*  xB  = (v16h*)(ws + 65536);                                  //   2 MB
    float* Rsw = (float*)(ws + 65536 + 2097152);                       //  64 MB
    float* pacc = Rsw + (size_t)S_ * S_;                               //  16 MB
    float* pl   = pacc + (size_t)J_ * B_ * S_ * D_;                    // 256 KB

    k_masses <<<B_ * S_ / 256, 256, 0, stream>>>(x, mw, mG, out_m);
    k_pack   <<<B_ * S_ * D_ / (16 * 256), 256, 0, stream>>>(x, xB);
    k_geo    <<<(S_ / 16) * (S_ / 32) * 32 / 256, 256, 0, stream>>>(pos, Rsw);
    k_attn   <<<B_ * J_ * 64, 128, 0, stream>>>(xB, Rsw, mG, G, pacc, pl);
    k_combine<<<B_ * S_ * D_ / 256, 256, 0, stream>>>(pacc, pl, out);
}
